// FeatureAttention_27075473834608
// MI455X (gfx1250) — compile-verified, matrix operand order checked by reading, not run on hardware
//
#include <hip/hip_runtime.h>
#include <stdint.h>

// ---------------------------------------------------------------------------
// Problem constants (from reference): B=32, E=16, F=64, D=512, EXP=4
// ---------------------------------------------------------------------------
#define F_ROWS 64
#define D_DIM  512
#define EXPD   2048
#define EPS_F  1e-5f
#define INV_SQRT_D 0.04419417382415922f  // 1/sqrt(512)

typedef __attribute__((ext_vector_type(16))) __bf16    v16bf;
typedef __attribute__((ext_vector_type(8)))  float     v8f;
typedef __attribute__((ext_vector_type(4)))  uint32_t  u32x4;

union FragB16 { v16bf v; u32x4 q[2]; };

#define WMMA_BF16(A, B, C) \
  __builtin_amdgcn_wmma_f32_16x16x32_bf16(false, (A), false, (B), (short)0, (C), false, false)

// Native f32 -> bf16 convert (single VALU op on gfx1250), bit-pattern out.
__device__ __forceinline__ uint16_t f32_to_bf16(float f) {
  union { __bf16 b; uint16_t u; } c;
  c.b = (__bf16)f;
  return c.u;
}

// Load one 16x32 bf16 WMMA operand fragment from a row-major bf16 buffer.
// A-matrix: row0+lane%16 = M row, K packed per ISA layout (half-wave K split).
// B-matrix: identical layout with row = N column of a pre-transposed weight.
// All strides used are even and rows are 16B aligned -> 128-bit loads.
__device__ __forceinline__ v16bf load_frag(const uint16_t* base, int stride,
                                           int row0, int k0, int lane) {
  const int r  = row0 + (lane & 15);
  const int kb = k0 + ((lane >> 4) << 3);
  const uint16_t* p = base + r * stride + kb;
  FragB16 f;
  f.q[0] = *(const u32x4*)(p);
  f.q[1] = *(const u32x4*)(p + 16);
  return f.v;
}

__device__ __forceinline__ float wave_sum(float v) {
#pragma unroll
  for (int off = 16; off > 0; off >>= 1) v += __shfl_xor(v, off, 32);
  return v;
}
__device__ __forceinline__ float wave_max(float v) {
#pragma unroll
  for (int off = 16; off > 0; off >>= 1) v = fmaxf(v, __shfl_xor(v, off, 32));
  return v;
}

// ---------------------------------------------------------------------------
// LDS arena (byte offsets). Peak = 301,056 B  <= 320 KB/WGP.
//   phases A-C : h (0..66559), q (66560..133119)
//   phase  E.. : x1 f32 overlays [0..133119]
//   k / h2     : 133120..199679
//   vT / ff1   : 199680..273407
//   energy     : 273408..291839   att: 291840..301055
// ---------------------------------------------------------------------------
#define OFF_Q    66560
#define OFF_K    133120
#define OFF_VT   199680
#define OFF_E    273408
#define OFF_ATT  291840
#define SMEM_BYTES 301056
#define LNS 520   // row stride (elements) for all D-wide [64][512] tiles (+8 pad)
#define LNV 72    // row stride for F-wide tiles (vT, energy, att)

// Weight convert + transpose: dst[n*K + k] = bf16(src[k*N + n])
__global__ void cvt_transpose_kernel(const float* __restrict__ src,
                                     uint16_t* __restrict__ dst, int K, int N) {
  int idx = blockIdx.x * blockDim.x + threadIdx.x;
  if (idx >= K * N) return;
  int k = idx / N, n = idx - k * N;
  dst[n * K + k] = f32_to_bf16(src[idx]);
}

__global__ __launch_bounds__(256, 1)
void feature_attn_kernel(const float* __restrict__ x,
                         const float* __restrict__ ln1_g, const float* __restrict__ ln1_b,
                         const float* __restrict__ bq, const float* __restrict__ bk,
                         const float* __restrict__ bv,
                         const float* __restrict__ wn_g, const float* __restrict__ wn_b,
                         const float* __restrict__ ln2_g, const float* __restrict__ ln2_b,
                         const float* __restrict__ b1, const float* __restrict__ b2,
                         const uint16_t* __restrict__ WqT, const uint16_t* __restrict__ WkT,
                         const uint16_t* __restrict__ WvT,
                         const uint16_t* __restrict__ W1T, const uint16_t* __restrict__ W2T,
                         float* __restrict__ out) {
  extern __shared__ char smem_raw[];
  uint16_t* sh_h   = (uint16_t*)(smem_raw);
  uint16_t* sh_q   = (uint16_t*)(smem_raw + OFF_Q);
  uint16_t* sh_k   = (uint16_t*)(smem_raw + OFF_K);
  uint16_t* sh_h2  = sh_k;                              // overlay (after phase C)
  uint16_t* sh_vt  = (uint16_t*)(smem_raw + OFF_VT);    // v transposed [512][72]
  uint16_t* sh_ff  = sh_vt;                             // overlay (FFN stage)
  float*    sh_e   = (float*)(smem_raw + OFF_E);        // energy [64][72]
  uint16_t* sh_att = (uint16_t*)(smem_raw + OFF_ATT);   // att [64][72]
  float*    sh_x1  = (float*)(smem_raw);                // overlay h+q (after phase D)

  const int tid  = threadIdx.x;
  const int lane = tid & 31;
  const int wv   = tid >> 5;                 // 8 wave32 per workgroup
  const int csub = lane & 15;                // C/D: n = lane%16
  const int mb   = (lane >> 4) << 3;         // C/D: m = vgpr + 8*(lane/16)
  const size_t gbase = (size_t)blockIdx.x * (F_ROWS * D_DIM);
  const v8f zero8 = {};                      // folds to WMMA inline-0 SRC2

  // ---- Phase A: LN1(x) -> h (bf16) -------------------------------------
  for (int r = 0; r < 8; ++r) {
    const int row = wv * 8 + r;
    const float* xr = x + gbase + (size_t)row * D_DIM;
    float v[16]; float s = 0.f;
#pragma unroll
    for (int j = 0; j < 16; ++j) { v[j] = xr[lane + j * 32]; s += v[j]; }
    const float mean = wave_sum(s) * (1.0f / 512.0f);
    float ss = 0.f;
#pragma unroll
    for (int j = 0; j < 16; ++j) { float d = v[j] - mean; ss += d * d; }
    const float rstd = rsqrtf(wave_sum(ss) * (1.0f / 512.0f) + EPS_F);
#pragma unroll
    for (int j = 0; j < 16; ++j) {
      const int c = lane + j * 32;
      sh_h[row * LNS + c] = f32_to_bf16((v[j] - mean) * rstd * ln1_g[c] + ln1_b[c]);
    }
  }
  __syncthreads();

  // ---- Phase B: q,k,v = h @ W{q,k,v}^T + b  (bf16 WMMA, f32 accum) -----
  // N-tiles processed in pairs so each A-fragment feeds 2 WMMAs (LDS reuse).
  // s-loop force-unrolled: destination + strides are compile-time constants.
  // kt=0 peeled so the first WMMA of each chain uses the inline-0 C operand.
#pragma unroll
  for (int s = 0; s < 3; ++s) {
    const uint16_t* WT  = (s == 0) ? WqT : (s == 1) ? WkT : WvT;
    const float*    bia = (s == 0) ? bq  : (s == 1) ? bk  : bv;
    uint16_t* dstb = (s == 2) ? sh_vt : (s == 0) ? sh_q : sh_k;
    const int sm   = (s == 2) ? 1   : LNS;   // stride applied to m index
    const int sn   = (s == 2) ? LNV : 1;     // stride applied to n index
    for (int jo = 0; jo < 2; ++jo) {
      const int nt0 = wv + 16 * jo;          // pair of N-tiles per wave
      const int nt1 = nt0 + 8;
      v8f acc[2][4];
      {  // peeled kt = 0: accumulators born from inline-0 C
        v16bf af[4];
#pragma unroll
        for (int mt = 0; mt < 4; ++mt) af[mt] = load_frag(sh_h, LNS, mt * 16, 0, lane);
        const v16bf bf0 = load_frag(WT, D_DIM, nt0 * 16, 0, lane);
#pragma unroll
        for (int mt = 0; mt < 4; ++mt) acc[0][mt] = WMMA_BF16(af[mt], bf0, zero8);
        const v16bf bf1 = load_frag(WT, D_DIM, nt1 * 16, 0, lane);
#pragma unroll
        for (int mt = 0; mt < 4; ++mt) acc[1][mt] = WMMA_BF16(af[mt], bf1, zero8);
      }
      for (int kt = 1; kt < 16; ++kt) {
        v16bf af[4];
#pragma unroll
        for (int mt = 0; mt < 4; ++mt) af[mt] = load_frag(sh_h, LNS, mt * 16, kt * 32, lane);
        const v16bf bf0 = load_frag(WT, D_DIM, nt0 * 16, kt * 32, lane);
#pragma unroll
        for (int mt = 0; mt < 4; ++mt) acc[0][mt] = WMMA_BF16(af[mt], bf0, acc[0][mt]);
        const v16bf bf1 = load_frag(WT, D_DIM, nt1 * 16, kt * 32, lane);
#pragma unroll
        for (int mt = 0; mt < 4; ++mt) acc[1][mt] = WMMA_BF16(af[mt], bf1, acc[1][mt]);
      }
#pragma unroll
      for (int p = 0; p < 2; ++p) {
        const int n = (p == 0 ? nt0 : nt1) * 16 + csub;
        const float bval = bia[n];
        uint16_t* drow = dstb + n * sn;      // fixed n-component of address
#pragma unroll
        for (int mt = 0; mt < 4; ++mt)
#pragma unroll
          for (int r2 = 0; r2 < 8; ++r2) {
            const int m = mt * 16 + mb + r2;
            drow[m * sm] = f32_to_bf16(acc[p][mt][r2] + bval);
          }
      }
    }
  }
  __syncthreads();

  // ---- Phase C: energy = q @ k^T  ([64,64], f32) -----------------------
  for (int t = wv * 2; t < wv * 2 + 2; ++t) {
    const int mt = t >> 2, nt = t & 3;
    v8f acc = WMMA_BF16(load_frag(sh_q, LNS, mt * 16, 0, lane),
                        load_frag(sh_k, LNS, nt * 16, 0, lane), zero8);
    for (int kt = 1; kt < 16; ++kt) {
      const v16bf af = load_frag(sh_q, LNS, mt * 16, kt * 32, lane);
      const v16bf bf = load_frag(sh_k, LNS, nt * 16, kt * 32, lane);  // row of k = key
      acc = WMMA_BF16(af, bf, acc);
    }
    const int n = nt * 16 + csub;
#pragma unroll
    for (int r2 = 0; r2 < 8; ++r2)
      sh_e[(mt * 16 + mb + r2) * LNV + n] = acc[r2];
  }
  __syncthreads();

  // ---- Phase D: LayerNorm over keys + softmax -> att (bf16) ------------
  for (int r = 0; r < 8; ++r) {
    const int row = wv * 8 + r;
    const float e0 = sh_e[row * LNV + lane];
    const float e1 = sh_e[row * LNV + lane + 32];
    const float mean = wave_sum(e0 + e1) * (1.0f / 64.0f);
    const float d0 = e0 - mean, d1 = e1 - mean;
    const float var = wave_sum(d0 * d0 + d1 * d1) * (1.0f / 64.0f);
    const float rstd = rsqrtf(var + EPS_F);
    float l0 = (d0 * rstd * wn_g[lane]      + wn_b[lane])      * INV_SQRT_D;
    float l1 = (d1 * rstd * wn_g[lane + 32] + wn_b[lane + 32]) * INV_SQRT_D;
    const float mx = wave_max(fmaxf(l0, l1));
    const float p0 = __expf(l0 - mx), p1 = __expf(l1 - mx);
    const float rs = 1.0f / wave_sum(p0 + p1);
    sh_att[row * LNV + lane]      = f32_to_bf16(p0 * rs);
    sh_att[row * LNV + lane + 32] = f32_to_bf16(p1 * rs);
  }
  __syncthreads();

  // ---- Phase E: x1 = x + att @ v  (f32, overlays h+q region) -----------
  for (int j = 0; j < 4; ++j) {
    const int nt = wv + 8 * j;
    v8f acc[4];
#pragma unroll
    for (int kt = 0; kt < 2; ++kt) {
      const v16bf bf = load_frag(sh_vt, LNV, nt * 16, kt * 32, lane);  // vT row = D col
#pragma unroll
      for (int mt = 0; mt < 4; ++mt) {
        const v16bf af = load_frag(sh_att, LNV, mt * 16, kt * 32, lane);
        acc[mt] = (kt == 0) ? WMMA_BF16(af, bf, zero8) : WMMA_BF16(af, bf, acc[mt]);
      }
    }
    const int n = nt * 16 + csub;
#pragma unroll
    for (int mt = 0; mt < 4; ++mt)
#pragma unroll
      for (int r2 = 0; r2 < 8; ++r2) {
        const int m = mt * 16 + mb + r2;
        sh_x1[m * LNS + n] = x[gbase + (size_t)m * D_DIM + n] + acc[mt][r2];
      }
  }
  __syncthreads();

  // ---- Phase F: LN2(x1) -> h2 (bf16) -----------------------------------
  for (int r = 0; r < 8; ++r) {
    const int row = wv * 8 + r;
    float v[16]; float s = 0.f;
#pragma unroll
    for (int j = 0; j < 16; ++j) { v[j] = sh_x1[row * LNS + lane + j * 32]; s += v[j]; }
    const float mean = wave_sum(s) * (1.0f / 512.0f);
    float ss = 0.f;
#pragma unroll
    for (int j = 0; j < 16; ++j) { float d = v[j] - mean; ss += d * d; }
    const float rstd = rsqrtf(wave_sum(ss) * (1.0f / 512.0f) + EPS_F);
#pragma unroll
    for (int j = 0; j < 16; ++j) {
      const int c = lane + j * 32;
      sh_h2[row * LNS + c] = f32_to_bf16((v[j] - mean) * rstd * ln2_g[c] + ln2_b[c]);
    }
  }
  __syncthreads();

  // ---- Phase G: FFN = (h2 @ W1 + b1) @ W2 + b2, K-blocked over 4 chunks -
  v8f acc2[4][4] = {};   // persistent GEMM2 accumulators: [nt-slot][mt] (128 f32/lane)
  for (int cb = 0; cb < 4; ++cb) {
    // GEMM1 block: ff1[:, cb*512 : cb*512+512] (bf16 into LDS), paired N-tiles
    for (int jo = 0; jo < 2; ++jo) {
      const int nt0 = wv + 16 * jo;
      const int nt1 = nt0 + 8;
      v8f acc[2][4];
      {  // peeled kt = 0
        v16bf af[4];
#pragma unroll
        for (int mt = 0; mt < 4; ++mt) af[mt] = load_frag(sh_h2, LNS, mt * 16, 0, lane);
        const v16bf bf0 = load_frag(W1T, D_DIM, cb * 512 + nt0 * 16, 0, lane);
#pragma unroll
        for (int mt = 0; mt < 4; ++mt) acc[0][mt] = WMMA_BF16(af[mt], bf0, zero8);
        const v16bf bf1 = load_frag(W1T, D_DIM, cb * 512 + nt1 * 16, 0, lane);
#pragma unroll
        for (int mt = 0; mt < 4; ++mt) acc[1][mt] = WMMA_BF16(af[mt], bf1, zero8);
      }
      for (int kt = 1; kt < 16; ++kt) {
        v16bf af[4];
#pragma unroll
        for (int mt = 0; mt < 4; ++mt) af[mt] = load_frag(sh_h2, LNS, mt * 16, kt * 32, lane);
        const v16bf bf0 = load_frag(W1T, D_DIM, cb * 512 + nt0 * 16, kt * 32, lane);
#pragma unroll
        for (int mt = 0; mt < 4; ++mt) acc[0][mt] = WMMA_BF16(af[mt], bf0, acc[0][mt]);
        const v16bf bf1 = load_frag(W1T, D_DIM, cb * 512 + nt1 * 16, kt * 32, lane);
#pragma unroll
        for (int mt = 0; mt < 4; ++mt) acc[1][mt] = WMMA_BF16(af[mt], bf1, acc[1][mt]);
      }
#pragma unroll
      for (int p = 0; p < 2; ++p) {
        const int nloc = (p == 0 ? nt0 : nt1) * 16 + csub;
        const float bval = b1[cb * 512 + nloc];
#pragma unroll
        for (int mt = 0; mt < 4; ++mt)
#pragma unroll
          for (int r2 = 0; r2 < 8; ++r2)
            sh_ff[(mt * 16 + mb + r2) * LNS + nloc] = f32_to_bf16(acc[p][mt][r2] + bval);
      }
    }
    __syncthreads();
    // GEMM2 partial: acc2 += ff1_cb @ W2[cb*512 : cb*512+512, :]
    // kt-outer: A-fragments loaded once per kt, reused by all 4 N-slots (16 WMMAs).
    for (int kt = 0; kt < 16; ++kt) {
      v16bf af[4];
#pragma unroll
      for (int mt = 0; mt < 4; ++mt) af[mt] = load_frag(sh_ff, LNS, mt * 16, kt * 32, lane);
#pragma unroll
      for (int j = 0; j < 4; ++j) {
        const int nt2 = wv + 8 * j;
        const v16bf bf = load_frag(W2T, EXPD, nt2 * 16, cb * 512 + kt * 32, lane);
#pragma unroll
        for (int mt = 0; mt < 4; ++mt)
          acc2[j][mt] = WMMA_BF16(af[mt], bf, acc2[j][mt]);
      }
    }
    __syncthreads();
  }

  // ---- Phase H: out = x1 + ff ------------------------------------------
  for (int j = 0; j < 4; ++j) {
    const int nt2 = wv + 8 * j;
    const int n = nt2 * 16 + csub;
    const float bval = b2[n];
#pragma unroll
    for (int mt = 0; mt < 4; ++mt)
#pragma unroll
      for (int r2 = 0; r2 < 8; ++r2) {
        const int m = mt * 16 + mb + r2;
        out[gbase + (size_t)m * D_DIM + n] = sh_x1[m * LNS + n] + acc2[j][mt][r2] + bval;
      }
  }
}

// ---------------------------------------------------------------------------
// Host launch. ws layout (bf16, transposed weights): WqT/WkT/WvT [512][512],
// W1T [2048][512], W2T [512][2048]  -> 5,767,168 bytes total.
// ---------------------------------------------------------------------------
extern "C" void kernel_launch(void* const* d_in, const int* in_sizes, int n_in,
                              void* d_out, int out_size, void* d_ws, size_t ws_size,
                              hipStream_t stream) {
  (void)in_sizes; (void)n_in; (void)out_size; (void)ws_size;
  const float* x     = (const float*)d_in[0];
  const float* ln1_g = (const float*)d_in[1];
  const float* ln1_b = (const float*)d_in[2];
  const float* Wq    = (const float*)d_in[3];
  const float* bq    = (const float*)d_in[4];
  const float* Wk    = (const float*)d_in[5];
  const float* bk    = (const float*)d_in[6];
  const float* Wv    = (const float*)d_in[7];
  const float* bv    = (const float*)d_in[8];
  const float* wn_g  = (const float*)d_in[9];
  const float* wn_b  = (const float*)d_in[10];
  const float* ln2_g = (const float*)d_in[11];
  const float* ln2_b = (const float*)d_in[12];
  const float* W1    = (const float*)d_in[13];
  const float* b1    = (const float*)d_in[14];
  const float* W2    = (const float*)d_in[15];
  const float* b2    = (const float*)d_in[16];

  uint16_t* WqT = (uint16_t*)d_ws;
  uint16_t* WkT = WqT + 512 * 512;
  uint16_t* WvT = WkT + 512 * 512;
  uint16_t* W1T = WvT + 512 * 512;
  uint16_t* W2T = W1T + 512 * 2048;

  cvt_transpose_kernel<<<(512 * 512) / 256, 256, 0, stream>>>(Wq, WqT, 512, 512);
  cvt_transpose_kernel<<<(512 * 512) / 256, 256, 0, stream>>>(Wk, WkT, 512, 512);
  cvt_transpose_kernel<<<(512 * 512) / 256, 256, 0, stream>>>(Wv, WvT, 512, 512);
  cvt_transpose_kernel<<<(512 * 2048) / 256, 256, 0, stream>>>(W1, W1T, 512, 2048);
  cvt_transpose_kernel<<<(512 * 2048) / 256, 256, 0, stream>>>(W2, W2T, 2048, 512);

  feature_attn_kernel<<<32 * 16, 256, SMEM_BYTES, stream>>>(
      x, ln1_g, ln1_b, bq, bk, bv, wn_g, wn_b, ln2_g, ln2_b, b1, b2,
      WqT, WkT, WvT, W1T, W2T, (float*)d_out);
}
